// Encoder_76192719831233
// MI455X (gfx1250) — compile-verified, matrix operand order checked by reading, not run on hardware
//
#include <hip/hip_runtime.h>
#include <hip/hip_bf16.h>
#include <math.h>
#include <stdint.h>

typedef __attribute__((ext_vector_type(16))) _Float16 v16h;
typedef __attribute__((ext_vector_type(8)))  float    v8f;

#define TPB 256

// ----------------------------------------------------------------------------
// Chunk loaders for the WMMA A/B operands (16-bit A 16x32 lane layout):
//   lanes 0-15 : elems 0..7 -> K kbase+0..7,  elems 8..15 -> K kbase+16..23
//   lanes 16-31: kbase = 8
// Fast path: 4 x float4 (global_load_b128 / ds_load_b128), unconditional.
// Guarded path: clamped address + value select (branchless, EXEC untouched).
// ----------------------------------------------------------------------------
__device__ __forceinline__ v16h load_chunk(const float* p, int k0) {
  const float4 x0 = *(const float4*)(p + k0);
  const float4 x1 = *(const float4*)(p + k0 + 4);
  const float4 x2 = *(const float4*)(p + k0 + 16);
  const float4 x3 = *(const float4*)(p + k0 + 20);
  v16h r;
  r[0]  = (_Float16)x0.x; r[1]  = (_Float16)x0.y; r[2]  = (_Float16)x0.z; r[3]  = (_Float16)x0.w;
  r[4]  = (_Float16)x1.x; r[5]  = (_Float16)x1.y; r[6]  = (_Float16)x1.z; r[7]  = (_Float16)x1.w;
  r[8]  = (_Float16)x2.x; r[9]  = (_Float16)x2.y; r[10] = (_Float16)x2.z; r[11] = (_Float16)x2.w;
  r[12] = (_Float16)x3.x; r[13] = (_Float16)x3.y; r[14] = (_Float16)x3.z; r[15] = (_Float16)x3.w;
  return r;
}

__device__ __forceinline__ v16h load_chunk_guard(const float* __restrict__ p,
                                                 int k0, int Kd) {
  v16h r;
#pragma unroll
  for (int i = 0; i < 16; ++i) {
    const int kk = k0 + i + (i & 8);          // K striping within chunk
    const int kc = (kk < Kd) ? kk : 0;        // clamp address (branchless)
    const float v = p[kc];                    // unconditional load
    r[i] = (_Float16)((kk < Kd) ? v : 0.0f);  // select, EXEC untouched
  }
  return r;
}

// ============================================================================
// WMMA GEMM:  C[M,O] = A[M,Kd] * W[O,Kd]^T + bias[O]
// Block = 8 waves = 128(M)x16(N) macro-tile. The 16xKd W tile is staged into
// LDS once per block with global_load_async_to_lds_b128 (ASYNCcnt), shared by
// all 8 waves; A rows stream from global. M must be a multiple of 128 so the
// grid is exact and EXEC stays all-ones for WMMA.
// ============================================================================
__global__ __launch_bounds__(256) void gemm_wmma_kernel(
    const float* __restrict__ A, const float* __restrict__ W,
    const float* __restrict__ bias, float* __restrict__ C,
    int M, int Kd, int O)
{
  __shared__ __align__(16) float wtile[16 * 512];   // up to Kd=512 (32 KB)

  const int tid  = threadIdx.x;
  const int lane = tid & 31;
  const int wave = tid >> 5;
  const int mblocks = M >> 7;                 // M / 128
  const int bm = blockIdx.x % mblocks;
  const int tn = blockIdx.x / mblocks;
  const int r15   = lane & 15;
  const int kbase = (lane >> 4) << 3;         // 0 or 8
  const int row0  = bm * 128 + wave * 16;
  const float* __restrict__ Ap = A + (long long)(row0 + r15) * Kd + kbase;

  v8f acc0 = {};
  if ((Kd & 3) == 0) {
    // ---- Stage W tile (16 contiguous rows = 16*Kd floats) into LDS ----
    const float* Wt = W + (long long)tn * 16 * Kd;
    const int nch = (16 * Kd) >> 2;           // # of 16-byte chunks
    for (int c = tid; c < nch; c += TPB) {
      const unsigned int laddr =
          (unsigned int)(uintptr_t)(&wtile[0]) + (unsigned int)c * 16u;
      const unsigned long long gaddr = (unsigned long long)(uintptr_t)(Wt + c * 4);
      asm volatile("global_load_async_to_lds_b128 %0, %1, off"
                   :: "v"(laddr), "v"(gaddr) : "memory");
    }
    asm volatile("s_wait_asynccnt 0x0" ::: "memory");
    __syncthreads();

    const float* Wl = &wtile[r15 * Kd + kbase];   // LDS, broadcast across waves
    v8f acc1 = {};
    const int kd64 = Kd & ~63;
    int k0 = 0;
    for (; k0 < kd64; k0 += 64) {
      __builtin_prefetch(Ap + k0 + 64, 0, 1);     // global_prefetch next A chunks
      acc0 = __builtin_amdgcn_wmma_f32_16x16x32_f16(
          false, load_chunk(Ap, k0), false, load_chunk(Wl, k0),
          (short)0, acc0, false, false);
      acc1 = __builtin_amdgcn_wmma_f32_16x16x32_f16(
          false, load_chunk(Ap, k0 + 32), false, load_chunk(Wl, k0 + 32),
          (short)0, acc1, false, false);
    }
    if (k0 < Kd) {
      acc0 = __builtin_amdgcn_wmma_f32_16x16x32_f16(
          false, load_chunk(Ap, k0), false, load_chunk(Wl, k0),
          (short)0, acc0, false, false);
    }
#pragma unroll
    for (int r = 0; r < 8; ++r) acc0[r] += acc1[r];
  } else {
    // Ragged Kd (transfer conv: 35/67/131/259): branchless clamp + select.
    const float* __restrict__ Wp = W + (long long)(tn * 16 + r15) * Kd + kbase;
    for (int k0 = 0; k0 < Kd; k0 += 32) {
      acc0 = __builtin_amdgcn_wmma_f32_16x16x32_f16(
          false, load_chunk_guard(Ap, k0, Kd - kbase),
          false, load_chunk_guard(Wp, k0, Kd - kbase),
          (short)0, acc0, false, false);
    }
  }

  // D layout: VGPR r -> row row0 + r (+8 for lanes 16-31), col = tn*16 + r15
  const int col  = tn * 16 + r15;
  const int rofs = (lane >> 4) << 3;
  const float bv = bias ? bias[col] : 0.0f;
#pragma unroll
  for (int r = 0; r < 8; ++r) {
    const int mrow = row0 + rofs + r;
    C[(long long)mrow * O + col] = acc0[r] + bv;
  }
}

// ============================================================================
// Per-output-channel mean/var over all rows (training-mode BatchNorm stats)
// ============================================================================
__global__ __launch_bounds__(256) void colstats_kernel(
    const float* __restrict__ X, long long M, int C,
    float* __restrict__ mean, float* __restrict__ var)
{
  const int c = blockIdx.x, tid = threadIdx.x;
  float s = 0.f, s2 = 0.f;
  for (long long m = tid; m < M; m += TPB) {
    const float v = X[m * C + c];
    s += v; s2 += v * v;
  }
  __shared__ float sh[TPB], sh2[TPB];
  sh[tid] = s; sh2[tid] = s2;
  __syncthreads();
  for (int st = TPB / 2; st; st >>= 1) {
    if (tid < st) { sh[tid] += sh[tid + st]; sh2[tid] += sh2[tid + st]; }
    __syncthreads();
  }
  if (tid == 0) {
    const float mu = sh[0] / (float)M;
    mean[c] = mu;
    var[c]  = sh2[0] / (float)M - mu * mu;
  }
}

// ============================================================================
// y = g*(x-m)*rsqrt(v+eps)+be  [+ residual] [ReLU]   (Y may alias X or res)
// ============================================================================
__global__ void bn_act_kernel(
    float* __restrict__ Y, const float* __restrict__ X,
    const float* __restrict__ mean, const float* __restrict__ var,
    const float* __restrict__ g, const float* __restrict__ be,
    const float* __restrict__ res, int relu, long long total, int C)
{
  const long long i = (long long)blockIdx.x * blockDim.x + threadIdx.x;
  if (i >= total) return;
  const int c = (int)(i % C);
  float v = g[c] * (X[i] - mean[c]) * rsqrtf(var[c] + 1e-5f) + be[c];
  if (res)  v += res[i];
  if (relu) v = fmaxf(v, 0.0f);
  Y[i] = v;
}

// ============================================================================
// Embed pointwise conv [B,3,N] -> [B*N,32]; also writes xyz_0 = transpose
// ============================================================================
__global__ void embed_pconv_kernel(
    const float* __restrict__ xyz, const float* __restrict__ W,
    const float* __restrict__ bias, float* __restrict__ out,
    float* __restrict__ xyz0, int B, int N)
{
  const long long i = (long long)blockIdx.x * blockDim.x + threadIdx.x;
  if (i >= (long long)B * N) return;
  const int b = (int)(i / N), n = (int)(i % N);
  const float x = xyz[((long long)b * 3 + 0) * N + n];
  const float y = xyz[((long long)b * 3 + 1) * N + n];
  const float z = xyz[((long long)b * 3 + 2) * N + n];
  xyz0[i * 3 + 0] = x; xyz0[i * 3 + 1] = y; xyz0[i * 3 + 2] = z;
#pragma unroll
  for (int o = 0; o < 32; ++o)
    out[i * 32 + o] = W[o * 3] * x + W[o * 3 + 1] * y + W[o * 3 + 2] * z + bias[o];
}

// ============================================================================
// Furthest point sampling: one workgroup per batch, dists in LDS, seq. argmax
// ============================================================================
__global__ __launch_bounds__(256) void fps_kernel(
    const float* __restrict__ xyz, int* __restrict__ idx, int N, int s)
{
  __shared__ float dist[4096];
  __shared__ float rv[TPB];
  __shared__ int   ri[TPB];
  __shared__ int   lastSh;
  const int b = blockIdx.x, tid = threadIdx.x;
  const float* X = xyz + (long long)b * N * 3;
  for (int n = tid; n < N; n += TPB) dist[n] = 1e10f;
  if (tid == 0) { idx[(long long)b * s] = 0; lastSh = 0; }
  __syncthreads();
  for (int it = 1; it < s; ++it) {
    const int last = lastSh;
    const float px = X[last * 3], py = X[last * 3 + 1], pz = X[last * 3 + 2];
    float bv = -1.f; int bi = 0;
    for (int n = tid; n < N; n += TPB) {
      const float dx = X[n * 3] - px, dy = X[n * 3 + 1] - py, dz = X[n * 3 + 2] - pz;
      const float dd = fminf(dist[n], dx * dx + dy * dy + dz * dz);
      dist[n] = dd;
      if (dd > bv) { bv = dd; bi = n; }
    }
    rv[tid] = bv; ri[tid] = bi;
    __syncthreads();
    for (int st = TPB / 2; st; st >>= 1) {
      if (tid < st) {
        if (rv[tid + st] > rv[tid] ||
            (rv[tid + st] == rv[tid] && ri[tid + st] < ri[tid])) {
          rv[tid] = rv[tid + st]; ri[tid] = ri[tid + st];
        }
      }
      __syncthreads();
    }
    if (tid == 0) { lastSh = ri[0]; idx[(long long)b * s + it] = ri[0]; }
    __syncthreads();
  }
}

// ============================================================================
// Gather sampled centers' xyz into output
// ============================================================================
__global__ void gather_xyz_kernel(
    const float* __restrict__ xyz, const int* __restrict__ idxc,
    float* __restrict__ outx, int s, int N, long long total)
{
  const long long i = (long long)blockIdx.x * blockDim.x + threadIdx.x;
  if (i >= total) return;
  const int d = (int)(i % 3);
  const long long t = i / 3;
  const int j = (int)(t % s), b = (int)(t / s);
  const int ci = idxc[(long long)b * s + j];
  outx[i] = xyz[((long long)b * N + ci) * 3 + d];
}

// ============================================================================
// kNN (K=32): one workgroup per center, d2 in LDS, 32 iterative min-extracts
// ============================================================================
__global__ __launch_bounds__(256) void knn_kernel(
    const float* __restrict__ cent, const float* __restrict__ xyz,
    int* __restrict__ nn, int s, int N)
{
  __shared__ float d2[4096];
  __shared__ float rv[TPB];
  __shared__ int   ri[TPB];
  const long long cs = blockIdx.x;
  const int b = (int)(cs / s), tid = threadIdx.x;
  const float* X = xyz + (long long)b * N * 3;
  const float cx = cent[cs * 3], cy = cent[cs * 3 + 1], cz = cent[cs * 3 + 2];
  for (int n = tid; n < N; n += TPB) {
    const float dx = X[n * 3] - cx, dy = X[n * 3 + 1] - cy, dz = X[n * 3 + 2] - cz;
    d2[n] = dx * dx + dy * dy + dz * dz;
  }
  __syncthreads();
  for (int k = 0; k < 32; ++k) {
    float bv = 3.4e38f; int bi = N;
    for (int n = tid; n < N; n += TPB) {
      const float v = d2[n];
      if (v < bv) { bv = v; bi = n; }
    }
    rv[tid] = bv; ri[tid] = bi;
    __syncthreads();
    for (int st = TPB / 2; st; st >>= 1) {
      if (tid < st) {
        if (rv[tid + st] < rv[tid] ||
            (rv[tid + st] == rv[tid] && ri[tid + st] < ri[tid])) {
          rv[tid] = rv[tid + st]; ri[tid] = ri[tid + st];
        }
      }
      __syncthreads();
    }
    if (tid == 0) { nn[cs * 32 + k] = ri[0]; d2[ri[0]] = 3.4e38f; }
    __syncthreads();
  }
}

// ============================================================================
// Grouping: G[b,j,k,c] = concat(f,xyz)[nn] - concat(f,xyz)[center]
// ============================================================================
__global__ void group_kernel(
    const float* __restrict__ f, const float* __restrict__ xyz,
    const int* __restrict__ idxc, const int* __restrict__ nn,
    float* __restrict__ G, int s, int N, int C, long long total)
{
  const long long i = (long long)blockIdx.x * blockDim.x + threadIdx.x;
  if (i >= total) return;
  const int Cp = C + 3;
  const int c = (int)(i % Cp);
  long long t = i / Cp;
  const int k = (int)(t & 31); t >>= 5;
  const int j = (int)(t % s), b = (int)(t / s);
  const long long bs = (long long)b * s + j;
  const int ni = nn[bs * 32 + k];
  const int ci = idxc[bs];
  float gv, av;
  if (c < C) {
    gv = f[((long long)b * N + ni) * C + c];
    av = f[((long long)b * N + ci) * C + c];
  } else {
    gv = xyz[((long long)b * N + ni) * 3 + (c - C)];
    av = xyz[((long long)b * N + ci) * 3 + (c - C)];
  }
  G[i] = gv - av;
}

// ============================================================================
// Per-batch scalar sum/sumsq over the grouped tensor (for jnp.std)
// ============================================================================
__global__ __launch_bounds__(256) void batch_stats_kernel(
    const float* __restrict__ G, float* __restrict__ stat, long long per)
{
  const int b = blockIdx.x, tid = threadIdx.x;
  const float* p = G + (long long)b * per;
  float s = 0.f, s2 = 0.f;
  for (long long i = tid; i < per; i += TPB) { const float v = p[i]; s += v; s2 += v * v; }
  __shared__ float sh[TPB], sh2[TPB];
  sh[tid] = s; sh2[tid] = s2;
  __syncthreads();
  for (int st = TPB / 2; st; st >>= 1) {
    if (tid < st) { sh[tid] += sh[tid + st]; sh2[tid] += sh2[tid + st]; }
    __syncthreads();
  }
  if (tid == 0) { stat[b * 2] = sh[0]; stat[b * 2 + 1] = sh2[0]; }
}

// ============================================================================
// Geometric affine: G = alpha[c] * G/(std_b + eps) + beta[c]
// ============================================================================
__global__ void affine_kernel(
    float* __restrict__ G, const float* __restrict__ stat,
    const float* __restrict__ alpha, const float* __restrict__ beta,
    long long per, int Cp, long long total)
{
  const long long i = (long long)blockIdx.x * blockDim.x + threadIdx.x;
  if (i >= total) return;
  const int b = (int)(i / per);
  const int c = (int)(i % Cp);
  const float n  = (float)per;
  const float mu = stat[b * 2] / n;
  const float va = stat[b * 2 + 1] / n - mu * mu;
  const float sd = sqrtf(fmaxf(va, 0.0f));
  G[i] = alpha[c] * (G[i] / (sd + 1e-5f)) + beta[c];
}

// ============================================================================
// Max-pool over the K=32 neighbor dim: [rows,32,O] -> [rows,O]
// ============================================================================
__global__ void maxpool_kernel(
    const float* __restrict__ H, float* __restrict__ P, int O, long long total)
{
  const long long i = (long long)blockIdx.x * blockDim.x + threadIdx.x;
  if (i >= total) return;
  const int o = (int)(i % O);
  const long long r = i / O;
  const float* p = H + (r * 32) * (long long)O + o;
  float m = p[0];
#pragma unroll
  for (int k = 1; k < 32; ++k) m = fmaxf(m, p[(long long)k * O]);
  P[i] = m;
}

// ============================================================================
// Host-side orchestration
// ============================================================================
struct RbP { const float *W1, *W2, *b1, *b2, *be1, *be2, *g1, *g2; };

static inline int cdiv_ll(long long a, int b) { return (int)((a + b - 1) / b); }

static void run_gemm(const float* A, const float* W, const float* b, float* C,
                     long long M, int Kd, int O, hipStream_t st) {
  const long long blocks = (M / 128) * (O / 16);   // M always multiple of 128 here
  gemm_wmma_kernel<<<(int)blocks, 256, 0, st>>>(A, W, b, C, (int)M, Kd, O);
}
static void run_stats(const float* X, long long M, int C, float* mean, float* var,
                      hipStream_t st) {
  colstats_kernel<<<C, 256, 0, st>>>(X, M, C, mean, var);
}
static void run_bn(float* Y, const float* X, const float* mean, const float* var,
                   const float* g, const float* be, const float* res, int relu,
                   long long M, int C, hipStream_t st) {
  const long long tot = M * C;
  bn_act_kernel<<<cdiv_ll(tot, 256), 256, 0, st>>>(Y, X, mean, var, g, be, res, relu, tot, C);
}
// res_block: out = relu( bn(conv2(relu(bn(conv1(in))))) + in )   (out may alias in)
static void run_res_block(const float* in, float* t1, float* t2, float* outp,
                          const RbP& p, long long M, int C,
                          float* mean, float* var, hipStream_t st) {
  run_gemm(in, p.W1, p.b1, t1, M, C, C, st);
  run_stats(t1, M, C, mean, var, st);
  run_bn(t1, t1, mean, var, p.g1, p.be1, nullptr, 1, M, C, st);
  run_gemm(t1, p.W2, p.b2, t2, M, C, C, st);
  run_stats(t2, M, C, mean, var, st);
  run_bn(outp, t2, mean, var, p.g2, p.be2, in, 1, M, C, st);
}
static inline RbP getRb(const float* const* P, int base) {
  return RbP{P[base], P[base + 1], P[base + 2], P[base + 3],
             P[base + 4], P[base + 5], P[base + 6], P[base + 7]};
}

extern "C" void kernel_launch(void* const* d_in, const int* in_sizes, int n_in,
                              void* d_out, int out_size, void* d_ws, size_t ws_size,
                              hipStream_t stream) {
  const int B = 8, N0 = 4096;

  // Locate xyz by its unique flat size; remaining leaves are params in
  // JAX sorted-key flatten order.
  int xyzIdx = -1;
  for (int i = 0; i < n_in; ++i)
    if (in_sizes[i] == B * 3 * N0) { xyzIdx = i; break; }
  if (xyzIdx < 0) xyzIdx = n_in - 1;
  const float* xyz_in = (const float*)d_in[xyzIdx];
  const float* P[200];
  int np = 0;
  for (int i = 0; i < n_in && np < 200; ++i)
    if (i != xyzIdx) P[np++] = (const float*)d_in[i];
  // embed (lin sorted keys: W, b, be, g)
  const float *eW = P[0], *eb = P[1], *ebe = P[2], *eg = P[3];

  // Output offsets (return order: xyz_0..xyz_4, f_0..f_4)
  float* out = (float*)d_out;
  const int Ns[5] = {4096, 2048, 1024, 512, 256};
  const int Cs[5] = {32, 64, 128, 256, 512};
  long long o_xyz[5], o_f[5], off = 0;
  for (int i = 0; i < 5; ++i) { o_xyz[i] = off; off += (long long)B * Ns[i] * 3; }
  for (int i = 0; i < 5; ++i) { o_f[i]   = off; off += (long long)B * Ns[i] * Cs[i]; }

  // Workspace carve-up (sizes constant across stages: s*K*O and s*O invariant)
  float* ws = (float*)d_ws;
  const long long HSZ = (long long)B * 2048 * 32 * 64;   // 33.6M floats
  const long long GSZ = (long long)B * 2048 * 32 * 35;   // max grouped tensor
  const long long PSZ = (long long)B * 2048 * 64;        // pooled / embed temp
  float* hA = ws;          float* hB = hA + HSZ;  float* hC = hB + HSZ;
  float* G  = hC + HSZ;
  float* pA = G + GSZ;     float* pB = pA + PSZ;  float* pC = pB + PSZ;
  float* mean = pC + PSZ;  float* var = mean + 512;  float* bstat = var + 512;
  int* idxb = (int*)(bstat + 16);
  int* nnb  = idxb + B * 2048;

  // ---- Embed: f_0 = relu(bn(pconv(pts))) ; xyz_0 = transpose(xyz) ----
  embed_pconv_kernel<<<cdiv_ll((long long)B * N0, 256), 256, 0, stream>>>(
      xyz_in, eW, eb, pA, out + o_xyz[0], B, N0);
  run_stats(pA, (long long)B * N0, 32, mean, var, stream);
  run_bn(out + o_f[0], pA, mean, var, eg, ebe, nullptr, 1, (long long)B * N0, 32, stream);

  // ---- Stages ----
  int Nin = N0, Cin = 32;
  for (int si = 0; si < 4; ++si) {
    const int s = Nin / 2, O = Cin * 2, Cp = Cin + 3;
    const int sb = 4 + 38 * si;   // stage keys sorted: alpha,beta,blocks1,blocks2,transfer
    const float* alpha = P[sb];
    const float* beta  = P[sb + 1];
    const RbP b10 = getRb(P, sb + 2),  b11 = getRb(P, sb + 10);
    const RbP b20 = getRb(P, sb + 18), b21 = getRb(P, sb + 26);
    const float *trW = P[sb + 34], *trb = P[sb + 35];
    const float *trbe = P[sb + 36], *trg = P[sb + 37];

    const float* curx = out + o_xyz[si];
    const float* fin  = out + o_f[si];
    float* outx = out + o_xyz[si + 1];
    float* fout = out + o_f[si + 1];

    // FPS -> centers -> kNN
    fps_kernel<<<B, 256, 0, stream>>>(curx, idxb, Nin, s);
    gather_xyz_kernel<<<cdiv_ll((long long)B * s * 3, 256), 256, 0, stream>>>(
        curx, idxb, outx, s, Nin, (long long)B * s * 3);
    knn_kernel<<<B * s, 256, 0, stream>>>(outx, curx, nnb, s, Nin);

    // Group + geometric affine
    const long long per = (long long)s * 32 * Cp;
    const long long gt  = (long long)B * per;
    group_kernel<<<cdiv_ll(gt, 256), 256, 0, stream>>>(fin, curx, idxb, nnb, G, s, Nin, Cin, gt);
    batch_stats_kernel<<<B, 256, 0, stream>>>(G, bstat, per);
    affine_kernel<<<cdiv_ll(gt, 256), 256, 0, stream>>>(G, bstat, alpha, beta, per, Cp, gt);

    // Transfer conv + BN + ReLU, then residual blocks over [B*s*K, O]
    const long long M1 = (long long)B * s * 32;
    run_gemm(G, trW, trb, hA, M1, Cp, O, stream);
    run_stats(hA, M1, O, mean, var, stream);
    run_bn(hA, hA, mean, var, trg, trbe, nullptr, 1, M1, O, stream);
    run_res_block(hA, hB, hC, hA, b10, M1, O, mean, var, stream);
    run_res_block(hA, hB, hC, hA, b11, M1, O, mean, var, stream);

    // Max-pool over K, then residual blocks over [B*s, O]
    const long long M2 = (long long)B * s;
    maxpool_kernel<<<cdiv_ll(M2 * O, 256), 256, 0, stream>>>(hA, pA, O, M2 * O);
    run_res_block(pA, pB, pC, pA, b20, M2, O, mean, var, stream);
    // last block writes its result directly into the f_{si+1} output slot
    run_gemm(pA, b21.W1, b21.b1, pB, M2, O, O, stream);
    run_stats(pB, M2, O, mean, var, stream);
    run_bn(pB, pB, mean, var, b21.g1, b21.be1, nullptr, 1, M2, O, stream);
    run_gemm(pB, b21.W2, b21.b2, pC, M2, O, O, stream);
    run_stats(pC, M2, O, mean, var, stream);
    run_bn(fout, pC, mean, var, b21.g2, b21.be2, pA, 1, M2, O, stream);

    Nin = s; Cin = O;
  }
  (void)out_size; (void)ws_size;
}
